// W2V2_RES2NET_COAASIST_61555471287014
// MI455X (gfx1250) — compile-verified, matrix operand order checked by reading, not run on hardware
//
#include <hip/hip_runtime.h>
#include <hip/hip_bf16.h>

typedef __attribute__((ext_vector_type(16))) _Float16 v16h;
typedef __attribute__((ext_vector_type(8)))  float    v8f;

#define BB 2
#define NN 512
#define DD 64
#define KNN 256
#define TEMP_INV (1.0f / 100.0f)

// Branch-free tanh: prefer the gfx1250 hardware v_tanh_f32.
__device__ __forceinline__ float fast_tanh(float x) {
#if __has_builtin(__builtin_amdgcn_tanhf)
    return __builtin_amdgcn_tanhf(x);
#else
    float ax = __builtin_fabsf(x);
    float e  = __expf(2.0f * ax);                    // v_exp_f32 path; inf on overflow
#if __has_builtin(__builtin_amdgcn_rcpf)
    float r  = __builtin_amdgcn_rcpf(e + 1.0f);      // v_rcp_f32, branch-free
#else
    float r  = 1.0f / (e + 1.0f);
#endif
    float t  = fmaf(-2.0f, r, 1.0f);                 // in [0,1]; e=inf -> 1
    return __builtin_copysignf(t, x);
#endif
}

// ---------------------------------------------------------------------------
// Kernel 1: one workgroup per (b,i).
//   scores_j = (1/T) * v . tanh( (x_j (*) x_i) @ W + b )    via WMMA f16
//   attn     = softmax_j(scores)
//   agg      = attn @ X
//   h_i      = agg @ Wpw + bpw + x_i @ Wpo + bpo            -> hbuf
// ---------------------------------------------------------------------------
__global__ __launch_bounds__(256) void attn_h_kernel(
    const float* __restrict__ x,    // (B,N,D)
    const float* __restrict__ Wp,   // att_proj_w (D,D)
    const float* __restrict__ bp,   // att_proj_b (D)
    const float* __restrict__ vw,   // att_weight (D,1)
    const float* __restrict__ Wpw,  // pw_att_w (D,D)
    const float* __restrict__ bpw,  // pw_att_b (D)
    const float* __restrict__ Wpo,  // pwo_att_w (D,D)
    const float* __restrict__ bpo,  // pwo_att_b (D)
    float* __restrict__ hbuf)       // (B*N, D)
{
    __shared__ __align__(16) float xi[DD];
    __shared__ __align__(16) float sc[NN];
    __shared__ float battn[DD];
    __shared__ float vsc[DD];
    __shared__ float red[256];
    __shared__ float agg[DD];

    const int bi   = blockIdx.x;          // 0..B*N-1
    const int b    = bi >> 9;
    const int tid  = threadIdx.x;
    const int lane = tid & 31;
    const int wave = tid >> 5;
    const int lm   = lane & 15;           // row (A) / col (B/C) within tile
    const int lg   = lane >> 4;           // K-half / M-half selector

    if (tid < DD) {
        xi[tid]    = x[(size_t)bi * DD + tid];
        battn[tid] = bp[tid];
        vsc[tid]   = vw[tid] * TEMP_INV;
    }
    __syncthreads();

    // ---- W fragments, B layout (mirror of A over (n,k)): barr[kt][nt] ----
    v16h barr[2][4];
#pragma unroll
    for (int kt = 0; kt < 2; ++kt) {
#pragma unroll
        for (int nt = 0; nt < 4; ++nt) {
            const int n = nt * 16 + lm;
#pragma unroll
            for (int p = 0; p < 8; ++p) {
                const int k = kt * 32 + (p >> 2) * 16 + lg * 8 + (p & 3) * 2;
                barr[kt][nt][2 * p]     = (_Float16)Wp[k * DD + n];
                barr[kt][nt][2 * p + 1] = (_Float16)Wp[(k + 1) * DD + n];
            }
        }
    }
    float bias_n[4], v_n[4];
#pragma unroll
    for (int nt = 0; nt < 4; ++nt) {
        bias_n[nt] = battn[nt * 16 + lm];
        v_n[nt]    = vsc[nt * 16 + lm];
    }

    // ---- per-wave: 4 M-tiles of 16 j-rows each ----
#pragma unroll 1
    for (int mm = 0; mm < 4; ++mm) {
        const int jbase = (wave * 4 + mm) * 16;
        const int j     = jbase + lm;
        const float* xr = x + ((size_t)(b * NN + j)) * DD;

        v16h a0, a1;   // K=0..31, K=32..63
#pragma unroll
        for (int g = 0; g < 2; ++g) {          // VGPR quad group: K base 0 / 16
            const int base = g * 16 + lg * 8;  // 8 contiguous K values
            float4 xa  = *(const float4*)(xr + base);
            float4 xb  = *(const float4*)(xr + base + 4);
            float4 ya  = *(const float4*)(&xi[base]);
            float4 yb  = *(const float4*)(&xi[base + 4]);
            a0[8 * g + 0] = (_Float16)(xa.x * ya.x);
            a0[8 * g + 1] = (_Float16)(xa.y * ya.y);
            a0[8 * g + 2] = (_Float16)(xa.z * ya.z);
            a0[8 * g + 3] = (_Float16)(xa.w * ya.w);
            a0[8 * g + 4] = (_Float16)(xb.x * yb.x);
            a0[8 * g + 5] = (_Float16)(xb.y * yb.y);
            a0[8 * g + 6] = (_Float16)(xb.z * yb.z);
            a0[8 * g + 7] = (_Float16)(xb.w * yb.w);
            float4 xc  = *(const float4*)(xr + 32 + base);
            float4 xd  = *(const float4*)(xr + 32 + base + 4);
            float4 yc  = *(const float4*)(&xi[32 + base]);
            float4 yd  = *(const float4*)(&xi[32 + base + 4]);
            a1[8 * g + 0] = (_Float16)(xc.x * yc.x);
            a1[8 * g + 1] = (_Float16)(xc.y * yc.y);
            a1[8 * g + 2] = (_Float16)(xc.z * yc.z);
            a1[8 * g + 3] = (_Float16)(xc.w * yc.w);
            a1[8 * g + 4] = (_Float16)(xd.x * yd.x);
            a1[8 * g + 5] = (_Float16)(xd.y * yd.y);
            a1[8 * g + 6] = (_Float16)(xd.z * yd.z);
            a1[8 * g + 7] = (_Float16)(xd.w * yd.w);
        }

        float part[8] = {0.f, 0.f, 0.f, 0.f, 0.f, 0.f, 0.f, 0.f};
#pragma unroll
        for (int nt = 0; nt < 4; ++nt) {
            v8f acc = {};
            acc = __builtin_amdgcn_wmma_f32_16x16x32_f16(
                false, a0, false, barr[0][nt], (short)0, acc, false, false);
            acc = __builtin_amdgcn_wmma_f32_16x16x32_f16(
                false, a1, false, barr[1][nt], (short)0, acc, false, false);
#pragma unroll
            for (int r = 0; r < 8; ++r) {
                float y = fast_tanh(acc[r] + bias_n[nt]);
                part[r] += y * v_n[nt];
            }
        }
        // reduce over the 16 lanes holding the same row group
#pragma unroll
        for (int r = 0; r < 8; ++r) {
            float v = part[r];
            v += __shfl_xor(v, 1, 16);
            v += __shfl_xor(v, 2, 16);
            v += __shfl_xor(v, 4, 16);
            v += __shfl_xor(v, 8, 16);
            part[r] = v;
        }
        if (lm == 0) {
#pragma unroll
            for (int r = 0; r < 8; ++r) sc[jbase + lg * 8 + r] = part[r];
        }
    }
    __syncthreads();

    // ---- softmax over j (unnormalized exp; fold 1/sum into agg) ----
    float m = -INFINITY;
    for (int j2 = tid; j2 < NN; j2 += 256) m = fmaxf(m, sc[j2]);
    red[tid] = m;
    __syncthreads();
    for (int s = 128; s > 0; s >>= 1) {
        if (tid < s) red[tid] = fmaxf(red[tid], red[tid + s]);
        __syncthreads();
    }
    const float mx = red[0];
    __syncthreads();

    float sum = 0.f;
    for (int j2 = tid; j2 < NN; j2 += 256) {
        float e = __expf(sc[j2] - mx);
        sc[j2]  = e;
        sum += e;
    }
    red[tid] = sum;
    __syncthreads();
    for (int s = 128; s > 0; s >>= 1) {
        if (tid < s) red[tid] += red[tid + s];
        __syncthreads();
    }
    const float inv = 1.0f / red[0];
    __syncthreads();

    // ---- agg = (attn @ X) * inv ----
    {
        const int d = tid & 63, grp = tid >> 6;
        float accv = 0.f;
        const int j0 = grp * 128;
        for (int j2 = j0; j2 < j0 + 128; ++j2)
            accv += sc[j2] * x[((size_t)(b * NN + j2)) * DD + d];
        red[tid] = accv;
        __syncthreads();
        if (grp == 0)
            agg[d] = (red[tid] + red[tid + 64] + red[tid + 128] + red[tid + 192]) * inv;
        __syncthreads();
    }

    // ---- h_i = agg @ Wpw + bpw + x_i @ Wpo + bpo ----
    if (tid < DD) {
        const int d = tid;
        float hv = bpw[d] + bpo[d];
        for (int k = 0; k < DD; ++k)
            hv += agg[k] * Wpw[k * DD + d] + xi[k] * Wpo[k * DD + d];
        hbuf[(size_t)bi * DD + d] = hv;
    }
}

// ---------------------------------------------------------------------------
// Kernel 2: batch-norm statistics over all B*N = 1024 rows, per feature.
// ---------------------------------------------------------------------------
__global__ __launch_bounds__(256) void bn_stats_kernel(
    const float* __restrict__ hbuf, float* __restrict__ mean, float* __restrict__ rstd)
{
    __shared__ float s1[256], s2[256];
    const int tid = threadIdx.x, d = tid & 63, grp = tid >> 6;
    float sum = 0.f, sq = 0.f;
    const int r0 = grp * 256;
    for (int r = r0; r < r0 + 256; ++r) {
        float v = hbuf[r * DD + d];
        sum += v;
        sq  += v * v;
    }
    s1[tid] = sum;
    s2[tid] = sq;
    __syncthreads();
    if (grp == 0) {
        float S  = s1[tid] + s1[tid + 64] + s1[tid + 128] + s1[tid + 192];
        float Q  = s2[tid] + s2[tid + 64] + s2[tid + 128] + s2[tid + 192];
        float mu = S * (1.0f / (BB * NN));
        float va = Q * (1.0f / (BB * NN)) - mu * mu;
        mean[d] = mu;
        rstd[d] = rsqrtf(va + 1e-5f);
    }
}

// ---------------------------------------------------------------------------
// Kernel 3: BN-normalize + SELU (in place) + pool gate w = sigmoid(h.pw + pb)
// ---------------------------------------------------------------------------
__global__ __launch_bounds__(64) void norm_selu_pool_kernel(
    float* __restrict__ hbuf, const float* __restrict__ mean,
    const float* __restrict__ rstd, const float* __restrict__ gamma,
    const float* __restrict__ beta, const float* __restrict__ pw,
    const float* __restrict__ pb, float* __restrict__ wbuf)
{
    __shared__ float red[64];
    const int row = blockIdx.x, d = threadIdx.x;
    float hv = hbuf[(size_t)row * DD + d];
    float hn = (hv - mean[d]) * rstd[d] * gamma[d] + beta[d];
    const float SELU_SC = 1.0507009873554805f;
    const float SELU_AL = 1.6732632423543772f;
    float ex = __expf(fminf(hn, 0.f)) - 1.0f;           // branch-free SELU
    float s  = hn > 0.f ? SELU_SC * hn : SELU_SC * SELU_AL * ex;
    hbuf[(size_t)row * DD + d] = s;
    red[d] = s * pw[d];
    __syncthreads();
    for (int st = 32; st > 0; st >>= 1) {
        if (d < st) red[d] += red[d + st];
        __syncthreads();
    }
    if (d == 0) wbuf[row] = 1.0f / (1.0f + __expf(-(red[0] + pb[0])));
}

// ---------------------------------------------------------------------------
// Kernel 4: per batch, bitonic-sort 512 (w, idx) pairs into jax top_k order
// (descending w, ties -> smaller idx first), gather top 256 rows of h*w.
// ---------------------------------------------------------------------------
__global__ __launch_bounds__(256) void topk_gather_kernel(
    const float* __restrict__ hbuf, const float* __restrict__ wbuf,
    float* __restrict__ out)
{
    __shared__ float kw[NN];
    __shared__ int   ki[NN];
    const int b = blockIdx.x, tid = threadIdx.x;
    for (int t = tid; t < NN; t += 256) {
        kw[t] = wbuf[b * NN + t];
        ki[t] = t;
    }
    __syncthreads();
    for (int k = 2; k <= NN; k <<= 1) {
        for (int j = k >> 1; j > 0; j >>= 1) {
            for (int t = tid; t < NN; t += 256) {
                const int xj = t ^ j;
                if (xj > t) {
                    const bool seg = ((t & k) == 0);  // segment in final order
                    float wa = kw[t], wb = kw[xj];
                    int   ia = ki[t], ib = ki[xj];
                    // xBeforeT under (descending w, ascending idx on ties)
                    const bool xBeforeT = (wb > wa) || (wb == wa && ib < ia);
                    if (xBeforeT == seg) {
                        kw[t] = wb;  kw[xj] = wa;
                        ki[t] = ib;  ki[xj] = ia;
                    }
                }
            }
            __syncthreads();
        }
    }
    for (int e = tid; e < KNN * DD; e += 256) {
        const int r = e >> 6, d = e & 63;
        const int src = ki[r];
        out[((size_t)(b * KNN + r)) * DD + d] =
            hbuf[((size_t)(b * NN + src)) * DD + d] * kw[r];
    }
}

// ---------------------------------------------------------------------------
extern "C" void kernel_launch(void* const* d_in, const int* in_sizes, int n_in,
                              void* d_out, int out_size, void* d_ws, size_t ws_size,
                              hipStream_t stream)
{
    const float* x   = (const float*)d_in[0];
    const float* Wp  = (const float*)d_in[1];
    const float* bp  = (const float*)d_in[2];
    const float* vw  = (const float*)d_in[3];
    const float* Wpw = (const float*)d_in[4];
    const float* bpw = (const float*)d_in[5];
    const float* Wpo = (const float*)d_in[6];
    const float* bpo = (const float*)d_in[7];
    const float* gam = (const float*)d_in[8];
    const float* bet = (const float*)d_in[9];
    const float* pw  = (const float*)d_in[10];
    const float* pb  = (const float*)d_in[11];
    float* out = (float*)d_out;

    float* hbuf = (float*)d_ws;                 // B*N*D = 65536 floats
    float* wbuf = hbuf + (size_t)BB * NN * DD;  // B*N   = 1024 floats
    float* mean = wbuf + BB * NN;               // 64
    float* rstd = mean + DD;                    // 64

    attn_h_kernel<<<BB * NN, 256, 0, stream>>>(x, Wp, bp, vw, Wpw, bpw, Wpo, bpo, hbuf);
    bn_stats_kernel<<<1, 256, 0, stream>>>(hbuf, mean, rstd);
    norm_selu_pool_kernel<<<BB * NN, 64, 0, stream>>>(hbuf, mean, rstd, gam, bet, pw, pb, wbuf);
    topk_gather_kernel<<<BB, 256, 0, stream>>>(hbuf, wbuf, out);
}
